// MixtureOfExperts_35579509080554
// MI455X (gfx1250) — matrix operand order repaired and checked by reading, not compiled, not run on hardware
//
#include <hip/hip_runtime.h>

#define HDIM 1024
#define IDIM 2048
#define NEXP 8
#define NTOK 16384   // B*S = 4*4096
#define XS_STRIDE 1048

typedef __attribute__((ext_vector_type(16))) __bf16 v16bf;
typedef __attribute__((ext_vector_type(8)))  float  v8f;

union FragU { uint4 q[2]; v16bf v; };

// A fragment (16x32 bf16): lane half 0 holds K = k0+{0..7, 16..23}, half 1 holds K = k0+{8..15, 24..31}.
// p points at the first 8 elements; the second 8 live 16 elements later.
static __device__ inline v16bf ld_fragA(const __bf16* p) {
  FragU f;
  f.q[0] = *(const uint4*)(p);
  f.q[1] = *(const uint4*)(p + 16);
  return f.v;
}
// B fragment (32x16 bf16): lane half 0 holds K = k0+{0..7, 16..23}, half 1 holds K = k0+{8..15, 24..31}.
static __device__ inline v16bf ld_fragB(const __bf16* p) {
  FragU f;
  f.q[0] = *(const uint4*)(p);
  f.q[1] = *(const uint4*)(p + 16);
  return f.v;
}

// ---------------- fp32 -> bf16 conversion of activations (coalesced) ----------------
__global__ __launch_bounds__(256) void moe_cvt_kernel(const float* __restrict__ x,
                                                      __bf16* __restrict__ xb, int n) {
  int i = (blockIdx.x * 256 + threadIdx.x) * 8;
  if (i >= n) return;
  const float4* p = (const float4*)(x + i);
  float4 a = p[0], b = p[1];
  union { __bf16 h[8]; uint4 q; } u;
  u.h[0] = (__bf16)a.x; u.h[1] = (__bf16)a.y; u.h[2] = (__bf16)a.z; u.h[3] = (__bf16)a.w;
  u.h[4] = (__bf16)b.x; u.h[5] = (__bf16)b.y; u.h[6] = (__bf16)b.z; u.h[7] = (__bf16)b.w;
  *(uint4*)(xb + i) = u.q;
}

// ---------------- weight transpose + convert: fp32 [R][C] -> bf16 [C][R] per expert ----------------
// grid = (C/32, R/32, E), block = 256
__global__ __launch_bounds__(256) void moe_tcvt_kernel(const float* __restrict__ in,
                                                       __bf16* __restrict__ outp,
                                                       int R, int C) {
  __shared__ float t[32][33];
  const size_t mat = (size_t)R * C;
  const float* in_e  = in   + (size_t)blockIdx.z * mat;
  __bf16*      out_e = outp + (size_t)blockIdx.z * mat;
  const int r0 = blockIdx.y * 32, c0 = blockIdx.x * 32;
  const int col  = threadIdx.x & 31;
  const int row8 = threadIdx.x >> 5;   // 0..7
#pragma unroll
  for (int p = 0; p < 4; p++) {
    int r = row8 + p * 8;
    t[r][col] = in_e[(size_t)(r0 + r) * C + (c0 + col)];
  }
  __syncthreads();
#pragma unroll
  for (int p = 0; p < 4; p++) {
    int cr = row8 + p * 8;
    out_e[(size_t)(c0 + cr) * R + (r0 + col)] = (__bf16)t[col][cr];
  }
}

// ---------------- router: one wave32 per token ----------------
__global__ __launch_bounds__(256) void moe_router_kernel(const float* __restrict__ x,
                                                         const float* __restrict__ gw,
                                                         int* __restrict__ cnt,
                                                         int* __restrict__ tok,
                                                         float* __restrict__ wt) {
  int gid  = blockIdx.x * 256 + threadIdx.x;
  int t    = gid >> 5;
  int lane = threadIdx.x & 31;
  if (t >= NTOK) return;

  float acc[NEXP];
#pragma unroll
  for (int e = 0; e < NEXP; e++) acc[e] = 0.f;

  const float* xp = x + (size_t)t * HDIM;
  for (int h = lane; h < HDIM; h += 32) {
    float xv = xp[h];
    const float* g = gw + h * NEXP;
#pragma unroll
    for (int e = 0; e < NEXP; e++) acc[e] += xv * g[e];
  }
#pragma unroll
  for (int off = 16; off > 0; off >>= 1) {
#pragma unroll
    for (int e = 0; e < NEXP; e++) acc[e] += __shfl_xor(acc[e], off, 32);
  }

  if (lane == 0) {
    float m = acc[0];
#pragma unroll
    for (int e = 1; e < NEXP; e++) m = fmaxf(m, acc[e]);
    float p[NEXP];
#pragma unroll
    for (int e = 0; e < NEXP; e++) p[e] = __expf(acc[e] - m);
    // softmax normalizer cancels in the renormalized top-2 weights: w1 = p1/(p1+p2)
    int i1 = 0;
#pragma unroll
    for (int e = 1; e < NEXP; e++) if (p[e] > p[i1]) i1 = e;
    int i2 = (i1 == 0) ? 1 : 0;
#pragma unroll
    for (int e = 0; e < NEXP; e++) if (e != i1 && p[e] > p[i2]) i2 = e;
    float denom = p[i1] + p[i2];
    int pos1 = atomicAdd(&cnt[i1], 1);
    tok[i1 * NTOK + pos1] = t;
    wt [i1 * NTOK + pos1] = p[i1] / denom;
    int pos2 = atomicAdd(&cnt[i2], 1);
    tok[i2 * NTOK + pos2] = t;
    wt [i2 * NTOK + pos2] = p[i2] / denom;
  }
}

// ---------------- fused SwiGLU FFN per 32-token expert tile ----------------
// grid = (NTOK/32, NEXP), block = 256 (8 wave32)
// Weights pre-converted to bf16 and transposed to [n][k] so every B fragment is
// two contiguous 16-byte loads per lane; each B fragment feeds two M-half WMMAs.
__global__ __launch_bounds__(256) void moe_ffn_kernel(const __bf16* __restrict__ xb,
                                                      const __bf16* __restrict__ wgT,
                                                      const __bf16* __restrict__ wuT,
                                                      const __bf16* __restrict__ wdT,
                                                      const int*   __restrict__ cnt,
                                                      const int*   __restrict__ tok,
                                                      const float* __restrict__ wtl,
                                                      float* __restrict__ out) {
  __shared__ __align__(16) __bf16 xs [32 * XS_STRIDE]; // x tile (32 x 1024), padded stride
  __shared__ __align__(16) float  ub [32 * 68];        // U exchange (f32)
  __shared__ __align__(16) __bf16 hsb[32 * 72];        // h = silu(g)*u tile (bf16)
  __shared__ int   toks[32];
  __shared__ float wts [32];

  const int e      = blockIdx.y;
  const int nvalid = cnt[e];
  const int base   = blockIdx.x * 32;
  if (base >= nvalid) return;

  const int tid  = threadIdx.x;
  const int wave = tid >> 5;
  const int lane = tid & 31;
  const int lrow = lane & 15;
  const int lhi  = lane >> 4;
  const int nsub = wave & 3;

  if (tid < 32) {
    int r = base + tid;
    if (r < nvalid) { toks[tid] = tok[e * NTOK + r]; wts[tid] = wtl[e * NTOK + r]; }
    else            { toks[tid] = -1;                wts[tid] = 0.f; }
  }
  __syncthreads();

  // stage x tile (32 x 1024 bf16), zero rows past count
  for (int idx = tid; idx < 32 * (HDIM / 8); idx += 256) {
    int r  = idx >> 7;
    int c8 = (idx & 127) << 3;
    int t  = toks[r];
    uint4 q = make_uint4(0u, 0u, 0u, 0u);
    if (t >= 0) q = *(const uint4*)(xb + (size_t)t * HDIM + c8);
    *(uint4*)(&xs[r * XS_STRIDE + c8]) = q;
  }
  __syncthreads();

  const __bf16* wgT_e = wgT + (size_t)e * HDIM * IDIM;
  const __bf16* wuT_e = wuT + (size_t)e * HDIM * IDIM;
  const __bf16* wdT_e = wdT + (size_t)e * IDIM * HDIM;

  v8f y0[8], y1[8];
#pragma unroll
  for (int j = 0; j < 8; j++) {
    y0[j] = (v8f){0.f,0.f,0.f,0.f,0.f,0.f,0.f,0.f};
    y1[j] = (v8f){0.f,0.f,0.f,0.f,0.f,0.f,0.f,0.f};
  }

  for (int c0 = 0; c0 < IDIM; c0 += 64) {        // I-chunk of 64 columns
    v8f g0 = (v8f){0.f,0.f,0.f,0.f,0.f,0.f,0.f,0.f};
    v8f g1 = (v8f){0.f,0.f,0.f,0.f,0.f,0.f,0.f,0.f};

    // gate (waves 0..3) / up (waves 4..7): G,U[32x64] = X[32x1024] @ W[:, c0:c0+64]
    const __bf16* bgu = ((wave < 4) ? wgT_e : wuT_e)
                      + (size_t)(c0 + nsub * 16 + lrow) * HDIM;
#pragma unroll 4
    for (int kp = 0; kp < HDIM; kp += 32) {
      v16bf a0 = ld_fragA(&xs[lrow * XS_STRIDE + kp + lhi * 8]);
      v16bf a1 = ld_fragA(&xs[(16 + lrow) * XS_STRIDE + kp + lhi * 8]);
      v16bf b  = ld_fragB(bgu + kp + lhi * 8);
      g0 = __builtin_amdgcn_wmma_f32_16x16x32_bf16(false, a0, false, b, (short)0, g0, false, false);
      g1 = __builtin_amdgcn_wmma_f32_16x16x32_bf16(false, a1, false, b, (short)0, g1, false, false);
    }

    // exchange U (waves 4..7) -> LDS, then waves 0..3 compute h = silu(G)*U
    if (wave >= 4) {
#pragma unroll
      for (int v = 0; v < 8; v++) {
        ub[(v + lhi * 8)      * 68 + nsub * 16 + lrow] = g0[v];
        ub[(v + lhi * 8 + 16) * 68 + nsub * 16 + lrow] = g1[v];
      }
    }
    __syncthreads();
    if (wave < 4) {
#pragma unroll
      for (int mh = 0; mh < 2; mh++) {
#pragma unroll
        for (int v = 0; v < 8; v++) {
          int   m  = v + lhi * 8 + 16 * mh;
          float gg = mh ? g1[v] : g0[v];
          float uu = ub[m * 68 + nsub * 16 + lrow];
          float hv = (gg / (1.f + __expf(-gg))) * uu;   // silu(g) * u
          hsb[m * 72 + nsub * 16 + lrow] = (__bf16)hv;
        }
      }
    }
    __syncthreads();

    // down-projection: Y[32 x 1024] += h[32 x 64] @ Wd[c0:c0+64, :]
#pragma unroll
    for (int k0 = 0; k0 < 64; k0 += 32) {
      v16bf a0 = ld_fragA(&hsb[lrow * 72 + k0 + lhi * 8]);
      v16bf a1 = ld_fragA(&hsb[(16 + lrow) * 72 + k0 + lhi * 8]);
#pragma unroll
      for (int j = 0; j < 8; j++) {
        const int n0 = (wave + 8 * j) * 16;
        v16bf b = ld_fragB(wdT_e + (size_t)(n0 + lrow) * IDIM + c0 + k0 + lhi * 8);
        y0[j] = __builtin_amdgcn_wmma_f32_16x16x32_bf16(false, a0, false, b, (short)0, y0[j], false, false);
        y1[j] = __builtin_amdgcn_wmma_f32_16x16x32_bf16(false, a1, false, b, (short)0, y1[j], false, false);
      }
    }
  }

  // epilogue: scatter-add with per-token combine weight
#pragma unroll
  for (int j = 0; j < 8; j++) {
    const int col = (wave + 8 * j) * 16 + lrow;
#pragma unroll
    for (int mh = 0; mh < 2; mh++) {
#pragma unroll
      for (int v = 0; v < 8; v++) {
        int m = v + lhi * 8 + 16 * mh;
        int t = toks[m];
        if (t >= 0) {
          float val = (mh ? y1[j][v] : y0[j][v]) * wts[m];
          atomicAdd(&out[(size_t)t * HDIM + col], val);
        }
      }
    }
  }
}

extern "C" void kernel_launch(void* const* d_in, const int* in_sizes, int n_in,
                              void* d_out, int out_size, void* d_ws, size_t ws_size,
                              hipStream_t stream) {
  (void)in_sizes; (void)n_in; (void)ws_size;
  const float* x  = (const float*)d_in[0];
  const float* gw = (const float*)d_in[1];
  const float* wg = (const float*)d_in[2];
  const float* wu = (const float*)d_in[3];
  const float* wd = (const float*)d_in[4];
  float* out = (float*)d_out;

  const size_t wmat = (size_t)HDIM * IDIM * NEXP * sizeof(__bf16);  // 32 MB per tensor

  // workspace layout
  char*   ws  = (char*)d_ws;
  __bf16* xb  = (__bf16*)ws;                      // 32 MB bf16 activations
  size_t  off = (size_t)NTOK * HDIM * 2;
  __bf16* wgT = (__bf16*)(ws + off); off += wmat; // transposed bf16 gate_proj [E][I][H]
  __bf16* wuT = (__bf16*)(ws + off); off += wmat; // transposed bf16 up_proj   [E][I][H]
  __bf16* wdT = (__bf16*)(ws + off); off += wmat; // transposed bf16 down_proj [E][H][I]
  int*    cnt = (int*)(ws + off);    off += 256;  // per-expert counts
  int*    tk  = (int*)(ws + off);    off += (size_t)NEXP * NTOK * 4;
  float*  wt  = (float*)(ws + off);

  hipMemsetAsync(d_out, 0, (size_t)out_size * sizeof(float), stream);
  hipMemsetAsync(cnt, 0, NEXP * sizeof(int), stream);

  int ncv = NTOK * HDIM;
  moe_cvt_kernel<<<ncv / 8 / 256, 256, 0, stream>>>(x, xb, ncv);
  // weight transpose+convert: fp32 [R][C] -> bf16 [C][R] per expert
  moe_tcvt_kernel<<<dim3(IDIM / 32, HDIM / 32, NEXP), 256, 0, stream>>>(wg, wgT, HDIM, IDIM);
  moe_tcvt_kernel<<<dim3(IDIM / 32, HDIM / 32, NEXP), 256, 0, stream>>>(wu, wuT, HDIM, IDIM);
  moe_tcvt_kernel<<<dim3(HDIM / 32, IDIM / 32, NEXP), 256, 0, stream>>>(wd, wdT, IDIM, HDIM);
  moe_router_kernel<<<(NTOK * 32) / 256, 256, 0, stream>>>(x, gw, cnt, tk, wt);
  moe_ffn_kernel<<<dim3(NTOK / 32, NEXP), 256, 0, stream>>>(xb, wgT, wuT, wdT, cnt, tk, wt, out);
}
